// ShapleyQMixer_63428077027892
// MI455X (gfx1250) — matrix operand order repaired and checked by reading, not compiled, not run on hardware
//
#include <hip/hip_runtime.h>

// ---------------------------------------------------------------------------
// ShapleyQMixer forward for MI455X (gfx1250, wave32, WMMA)
//   B=32 T=32 D=256 N=16 S=16 E=64 ; bs = B*T = 1024
//   Pipeline:
//     k_stats : per-column RunningMeanStd update (mean / rstd) over 1024 rows
//     k_norm  : rsn -> f16 activation matrix A [1024 x 256]
//     k_pack  : concat W1|Wb1|Wf|Vw1 -> f16 Wcat, stored pre-swizzled in the
//               exact WMMA B-operand lane layout (32B contiguous per lane)
//     k_gemm  : WMMA f16 GEMM 1024x320x256 + fused |.| / id / relu epilogue
//     k_final : Shapley prefix sums, elu hidden layer, wf dot, V head, q_tot
// ---------------------------------------------------------------------------

typedef _Float16 v16h __attribute__((ext_vector_type(16)));
typedef float    v8f  __attribute__((ext_vector_type(8)));

#define BS   1024
#define DD   256
#define NN   16
#define SS   16
#define EE   64
#define NCAT 320   // 128 (w1) + 64 (b1h) + 64 (wf) + 64 (vh)
#define NTIL 20    // NCAT / 16
#define KSTP 8     // DD / 32

// workspace byte offsets (32B aligned where vector loads happen)
#define OFF_MEAN 0
#define OFF_RSTD 1024
#define OFF_BIAS 2048
#define OFF_A    3328                       // 1024*256 f16 = 524288 B
#define OFF_W    (OFF_A + 524288)           // 256*320 f16  = 163840 B (swizzled)
#define OFF_H    (OFF_W + 163840)           // 1024*320 f32 = 1310720 B

// ---- 1) column stats: RunningMeanStd.update from (mean=0,var=1,count=1e-4)
__global__ void __launch_bounds__(256)
k_stats(const float* __restrict__ states, float* __restrict__ mean,
        float* __restrict__ rstd) {
  const int c = blockIdx.x;        // column 0..255
  const int t = threadIdx.x;       // 0..255
  float s = 0.f, s2 = 0.f;
  for (int r = t; r < BS; r += 256) {
    float x = states[r * DD + c];
    s += x; s2 += x * x;
  }
  __shared__ float sh[256], sh2[256];
  sh[t] = s; sh2[t] = s2;
  __syncthreads();
  for (int off = 128; off > 0; off >>= 1) {
    if (t < off) { sh[t] += sh[t + off]; sh2[t] += sh2[t + off]; }
    __syncthreads();
  }
  if (t == 0) {
    const float n = (float)BS;
    float bm = sh[0] / n;
    float bv = (sh2[0] - n * bm * bm) / (n - 1.f);   // unbiased (ddof=1)
    float tot = 1e-4f + n;
    float nm  = bm * n / tot;
    float M2  = 1e-4f + bv * n + bm * bm * 1e-4f * n / tot;
    float nv  = M2 / tot;
    mean[c] = nm;
    rstd[c] = rsqrtf(nv);
  }
}

// ---- 2) normalize + convert to f16
__global__ void __launch_bounds__(256)
k_norm(const float* __restrict__ states, const float* __restrict__ mean,
       const float* __restrict__ rstd, _Float16* __restrict__ A) {
  int idx = blockIdx.x * 256 + threadIdx.x;   // 0..262143
  int c = idx & (DD - 1);
  A[idx] = (_Float16)((states[idx] - mean[c]) * rstd[c]);
}

// ---- 3) pack concatenated hypernet weights into the WMMA B-operand layout.
//  Logical Wcat[k][n], k<256, n<320. B 32x16 f16 tile (tn, ks), contiguous k order as in A:
//    lane = 16*(kr>=16) + (n&15), element j -> k = ks*32 + 16*(kr>=16) + j.
//  Swizzled store: Wcs[ ((tn*KSTP + ks)*32 + lane)*16 + j ]  (32B per lane).
__global__ void __launch_bounds__(256)
k_pack(const float* __restrict__ W1,  const float* __restrict__ b1,
       const float* __restrict__ Wb1, const float* __restrict__ bb1,
       const float* __restrict__ Wf,  const float* __restrict__ bf,
       const float* __restrict__ Vw1, const float* __restrict__ Vb1,
       _Float16* __restrict__ Wcs, float* __restrict__ bias) {
  int n = blockIdx.x;    // 0..319
  int k = threadIdx.x;   // 0..255
  float w, bb;
  if (n < 128)      { w = W1 [k * 128 + n];        bb = b1 [n];       }
  else if (n < 192) { w = Wb1[k * 64 + (n - 128)]; bb = bb1[n - 128]; }
  else if (n < 256) { w = Wf [k * 64 + (n - 192)]; bb = bf [n - 192]; }
  else              { w = Vw1[k * 64 + (n - 256)]; bb = Vb1[n - 256]; }
  const int tn   = n >> 4;          // n tile
  const int m    = n & 15;          // column within tile
  const int ks   = k >> 5;          // k step
  const int kr   = k & 31;          // k within step
  const int hi   = kr >> 4;         // half selector
  const int j    = kr & 15;         // element within lane
  const int lane = hi * 16 + m;
  Wcs[(((tn * KSTP + ks) * 32) + lane) * 16 + j] = (_Float16)w;
  if (k == 0) bias[n] = bb;
}

// ---- 4) WMMA GEMM: H[1024 x 320] = epilogue(A[1024x256] @ Wcat[256x320] + bias)
//      64 x 20 tiles of 16x16, one wave per tile, 8 waves per block.
__global__ void __launch_bounds__(256)
k_gemm(const _Float16* __restrict__ A, const _Float16* __restrict__ Wcs,
       const float* __restrict__ bias, float* __restrict__ H) {
  const int tid  = threadIdx.x;
  const int lane = tid & 31;
  const int wave = tid >> 5;
  const int gt   = blockIdx.x * 8 + wave;   // 0..1279
  const int tm   = gt / NTIL;               // 0..63  (M tile)
  const int tn   = gt % NTIL;               // 0..19  (N tile)
  const int m    = lane & 15;
  const int hi   = lane >> 4;               // 0 or 1
  const int row  = tm * 16 + m;             // A row for this lane
  const int col  = tn * 16 + m;             // B/D column for this lane

  const _Float16* pa0 = A + row * DD + hi * 16;
  const v16h* pb0 = (const v16h*)Wcs + (tn * KSTP * 32 + lane);

  v8f acc = {};
  #pragma unroll
  for (int ks = 0; ks < KSTP; ++ks) {
    // A operand, contiguous k order (same as B): lane<16 -> K {0..15}; lane>=16 -> K {16..31}
    const _Float16* pa = pa0 + ks * 32;
    v16h a;
    #pragma unroll
    for (int j = 0; j < 8; ++j) { a[j] = pa[j]; a[j + 8] = pa[8 + j]; }
    // B operand: pre-swizzled, one contiguous 32B vector per lane per step
    const v16h bvec = pb0[ks * 32];
    acc = __builtin_amdgcn_wmma_f32_16x16x32_f16(
        /*neg_a=*/false, a, /*neg_b=*/false, bvec,
        /*c_mod=*/(short)0, acc, /*reuse_a=*/false, /*reuse_b=*/false);
  }

  const float bcol = bias[col];
  #pragma unroll
  for (int r = 0; r < 8; ++r) {
    const int rr = tm * 16 + r + hi * 8;    // D row: lanes>=16 hold M=r+8
    float v = acc[r] + bcol;
    float o;
    if (col < 128)      o = fabsf(v);       // w1   = |.|
    else if (col < 192) o = v;              // b1h  = id
    else if (col < 256) o = fabsf(v);       // wf   = |.|
    else                o = fmaxf(v, 0.f);  // vh   = relu
    H[rr * NCAT + col] = o;
  }
}

// ---- 5) Shapley sampling + hidden layer + output mix. One block per bs row.
__global__ void __launch_bounds__(256)
k_final(const float* __restrict__ aq, const float* __restrict__ mf,
        const int* __restrict__ gc, const float* __restrict__ H,
        const float* __restrict__ Vw2, const float* __restrict__ Vb2,
        float* __restrict__ out) {
  const int b = blockIdx.x;    // 0..1023
  const int t = threadIdx.x;   // 0..255 -> (s = t>>4, i = t&15)
  __shared__ float aq_sh[NN];
  __shared__ int   gc_sh[SS * NN];
  __shared__ float hh[NCAT];
  __shared__ float v_sh;
  __shared__ float ys[256];
  __shared__ float wr[NN];

  if (t < NN) aq_sh[t] = aq[b * NN + t];
  gc_sh[t] = gc[b * SS * NN + t];
  hh[t] = H[b * NCAT + t];
  if (t < 64) hh[256 + t] = H[b * NCAT + 256 + t];
  __syncthreads();

  if (t == 0) {                        // V head: v = relu(.)@Vw2 + Vb2
    float v = Vb2[0];
    for (int e = 0; e < EE; ++e) v += hh[256 + e] * Vw2[e];
    v_sh = v;
  }
  __syncthreads();

  const int s = t >> 4, i = t & 15;
  // sub_no_i[b,s,i,j] == (j < gc[b,s,i]) -> csum = prefix sum of permuted q's
  const int pos = gc_sh[s * NN + i];
  float csum = 0.f;
  for (int j = 0; j < pos; ++j) csum += aq_sh[gc_sh[s * NN + j]];
  const float denom = (pos == 0) ? 1.f : (float)pos;
  const float nv = csum / denom;
  const float x1 = aq_sh[i];

  float y = 0.f;
  for (int e = 0; e < EE; ++e) {
    float z = nv * hh[e] + x1 * hh[64 + e] + hh[128 + e];
    float h = (z > 0.f) ? z : (expf(z) - 1.f);   // elu
    y += h * hh[192 + e];
  }
  y += v_sh;
  ys[t] = fabsf(y);
  __syncthreads();

  if (t < NN) {                        // w_est[b,i] = mean_s |y| + 1
    float w = 0.f;
    for (int ss = 0; ss < SS; ++ss) w += ys[ss * NN + t];
    w = w * (1.f / (float)SS) + 1.f;
    out[BS + b * NN + t] = w;          // w_est after the 1024 q_tot floats
    float m = mf[b * NN + t];
    wr[t] = (w * (1.f - m) + m) * aq_sh[t];
  }
  __syncthreads();
  if (t == 0) {
    float q = 0.f;
    for (int k = 0; k < NN; ++k) q += wr[k];
    out[b] = q;                        // q_tot
  }
}

extern "C" void kernel_launch(void* const* d_in, const int* in_sizes, int n_in,
                              void* d_out, int out_size, void* d_ws, size_t ws_size,
                              hipStream_t stream) {
  (void)in_sizes; (void)n_in; (void)out_size; (void)ws_size;
  const float* states     = (const float*)d_in[0];
  /* d_in[1] = actions, unused */
  const float* agent_qs   = (const float*)d_in[2];
  const float* max_filter = (const float*)d_in[3];
  const int*   gcoal      = (const int*)  d_in[4];
  const float* W1  = (const float*)d_in[5];
  const float* b1  = (const float*)d_in[6];
  const float* Wb1 = (const float*)d_in[7];
  const float* bb1 = (const float*)d_in[8];
  const float* Wf  = (const float*)d_in[9];
  const float* bf  = (const float*)d_in[10];
  const float* Vw1 = (const float*)d_in[11];
  const float* Vb1 = (const float*)d_in[12];
  const float* Vw2 = (const float*)d_in[13];
  const float* Vb2 = (const float*)d_in[14];

  char* ws = (char*)d_ws;
  float*    mean = (float*)(ws + OFF_MEAN);
  float*    rstd = (float*)(ws + OFF_RSTD);
  float*    bias = (float*)(ws + OFF_BIAS);
  _Float16* A    = (_Float16*)(ws + OFF_A);
  _Float16* Wcs  = (_Float16*)(ws + OFF_W);
  float*    H    = (float*)(ws + OFF_H);
  float*    out  = (float*)d_out;

  k_stats<<<DD, 256, 0, stream>>>(states, mean, rstd);
  k_norm <<<BS * DD / 256, 256, 0, stream>>>(states, mean, rstd, A);
  k_pack <<<NCAT, 256, 0, stream>>>(W1, b1, Wb1, bb1, Wf, bf, Vw1, Vb1, Wcs, bias);
  k_gemm <<<160, 256, 0, stream>>>(A, Wcs, bias, H);   // 160*8 waves = 64*20 tiles
  k_final<<<BS, 256, 0, stream>>>(agent_qs, max_filter, gcoal, H, Vw2, Vb2, out);
}
